// nonlinearMLP_57896159150711
// MI455X (gfx1250) — matrix operand order repaired and checked by reading, not compiled, not run on hardware
//
#include <hip/hip_runtime.h>
#include <hip/hip_bf16.h>

// MI455X / gfx1250, wave32 only.
typedef __attribute__((ext_vector_type(16))) _Float16 v16h;
typedef __attribute__((ext_vector_type(8)))  _Float16 v8h;
typedef __attribute__((ext_vector_type(8)))  float    v8f;
typedef __attribute__((ext_vector_type(4)))  float    v4f;

#define NN   256   // nodes
#define BB   1024  // batch
#define W1S  264   // padded LDS row stride (halves): 528B -> conflict-free banks

__device__ __forceinline__ float fast_tanh(float x) {
#if defined(__has_builtin) && __has_builtin(__builtin_amdgcn_tanhf)
  return __builtin_amdgcn_tanhf(x);
#else
  float r;
  asm volatile("v_tanh_f32 %0, %1" : "=v"(r) : "v"(x));
  return r;
#endif
}

__device__ __forceinline__ v8f wmma_f16(v16h a, v16h b, v8f c) {
  return __builtin_amdgcn_wmma_f32_16x16x32_f16(
      false, a, false, b, (short)0, c, false, false);
}

// Build the 8 A-fragments (16x32 f16 WMMA layout) for one 16-row strip:
//   h1[k] = tanh(base_row[k] - xv * w0col[k])
__device__ __forceinline__ void build_afrags(
    const float* __restrict__ base_row, float xv,
    const float* __restrict__ w0col, int half, v16h* af) {
#pragma unroll
  for (int kb = 0; kb < 8; ++kb) {
    const int koff = kb * 32 + half * 8;
    const v4f* bp = (const v4f*)(base_row + koff);
    v4f c0 = bp[0], c1 = bp[1], c2 = bp[4], c3 = bp[5];
    v16h a;
#pragma unroll
    for (int t = 0; t < 4; ++t) {
      a[t]      = (_Float16)fast_tanh(c0[t] - xv * w0col[koff + t]);
      a[t + 4]  = (_Float16)fast_tanh(c1[t] - xv * w0col[koff + 4 + t]);
      a[t + 8]  = (_Float16)fast_tanh(c2[t] - xv * w0col[koff + 16 + t]);
      a[t + 12] = (_Float16)fast_tanh(c3[t] - xv * w0col[koff + 20 + t]);
    }
    af[kb] = a;
  }
}

// ---------------------------------------------------------------------------
// K1: base[b,m] = sum_k x[b,k] * W0[m,k]   (shared layer-1 GEMM, f32 out)
// grid: 64 blocks x 256 threads. Global wave id gw in [0,512):
//   strip = gw & 63 (16 batch rows), quarter = gw >> 6 -> N-tile pair.
// ---------------------------------------------------------------------------
__global__ __launch_bounds__(256) void base_gemm_kernel(
    const float* __restrict__ x, const float* __restrict__ W0,
    float* __restrict__ base)
{
  const int lane    = threadIdx.x & 31;
  const int wave    = threadIdx.x >> 5;
  const int gw      = blockIdx.x * 8 + wave;
  const int strip   = gw & 63;
  const int quarter = gw >> 6;
  const int l16     = lane & 15;
  const int half    = lane >> 4;
  const int mrow    = strip * 16 + l16;

  // A fragments straight from x (f32 -> f16), kept in registers.
  v16h afrag[8];
#pragma unroll
  for (int kb = 0; kb < 8; ++kb) {
    const int koff = kb * 32 + half * 8;
    const v4f* xp = (const v4f*)(x + mrow * NN + koff);
    v4f c0 = xp[0], c1 = xp[1], c2 = xp[4], c3 = xp[5];
    v16h a;
#pragma unroll
    for (int t = 0; t < 4; ++t) {
      a[t]      = (_Float16)c0[t];
      a[t + 4]  = (_Float16)c1[t];
      a[t + 8]  = (_Float16)c2[t];
      a[t + 12] = (_Float16)c3[t];
    }
    afrag[kb] = a;
  }

#pragma unroll
  for (int nbi = 0; nbi < 2; ++nbi) {
    const int nb = quarter * 2 + nbi;
    const int n  = nb * 16 + l16;          // output column m of base
    v8f acc = {};
#pragma unroll
    for (int kb = 0; kb < 8; ++kb) {
      const int k0 = kb * 32 + half * 8;   // B: lanes<16 -> K=0..7,16..23 of block
      const v4f* wp = (const v4f*)(W0 + n * NN + k0);
      v4f w0v = wp[0], w1v = wp[1], w2v = wp[4], w3v = wp[5];
      v16h b;
#pragma unroll
      for (int t = 0; t < 4; ++t) {
        b[t]      = (_Float16)w0v[t];
        b[t + 4]  = (_Float16)w1v[t];
        b[t + 8]  = (_Float16)w2v[t];
        b[t + 12] = (_Float16)w3v[t];
      }
      acc = wmma_f16(afrag[kb], b, acc);
    }
    // C layout: VGPR r -> M = r + 8*half (per 16-lane group), N = lane%16
#pragma unroll
    for (int r = 0; r < 8; ++r) {
      const int m = strip * 16 + r + half * 8;
      base[m * NN + n] = acc[r];
    }
  }
}

// ---------------------------------------------------------------------------
// K2: one workgroup per zeroed column i.
//   h1 = tanh(base - x[:,i] * W0[:,i]^T)   (rank-1 corrected, built in regs)
//   h2 = tanh(h1 @ W1^T)                   (WMMA, W1 f16 staged in padded LDS)
//   out[:,i] = tanh(h2 @ W2[i,:])          (fused dot + lane butterfly)
// Two 16-row batch strips per wave iteration: each B fragment loaded from LDS
// once feeds TWO WMMAs (ds:wmma = 1:1) to stay under the LDS bandwidth roof.
// ---------------------------------------------------------------------------
__global__ __launch_bounds__(256, 1) void mlp_masked_kernel(
    const float* __restrict__ x,  const float* __restrict__ W0,
    const float* __restrict__ W1, const float* __restrict__ W2,
    const float* __restrict__ base, float* __restrict__ out)
{
  extern __shared__ char smem[];
  _Float16* w1h   = (_Float16*)smem;                   // 256 x 264 halves
  float*    w0col = (float*)(smem + NN * W1S * 2);     // W0[:, i]
  float*    w2row = w0col + NN;                        // W2[i, :]

  const int i    = blockIdx.x;
  const int tid  = threadIdx.x;
  const int lane = tid & 31;
  const int wave = tid >> 5;
  const int l16  = lane & 15;
  const int half = lane >> 4;

  // Stage W1 (f32 -> f16) into padded LDS; W0 column i; W2 row i.
  for (int e = tid; e < NN * NN; e += 256) {
    const int r = e >> 8, c = e & 255;
    w1h[r * W1S + c] = (_Float16)W1[e];
  }
  if (tid < NN) {
    w0col[tid] = W0[tid * NN + i];
    w2row[tid] = W2[i * NN + tid];
  }
  __syncthreads();

  for (int sp = wave; sp < BB / 32; sp += 8) {   // 32 double-strips
    const int s0 = sp * 2, s1 = sp * 2 + 1;
    const int mrow0 = s0 * 16 + l16;
    const int mrow1 = s1 * 16 + l16;
    const float xv0 = x[mrow0 * NN + i];
    const float xv1 = x[mrow1 * NN + i];

    v16h afA[8], afB[8];
    build_afrags(base + mrow0 * NN, xv0, w0col, half, afA);
    build_afrags(base + mrow1 * NN, xv1, w0col, half, afB);

    float dotA[8] = {0.f, 0.f, 0.f, 0.f, 0.f, 0.f, 0.f, 0.f};
    float dotB[8] = {0.f, 0.f, 0.f, 0.f, 0.f, 0.f, 0.f, 0.f};

#pragma unroll 2
    for (int nb = 0; nb < 16; ++nb) {
      const int n = nb * 16 + l16;
      v8f accA = {}, accB = {};
#pragma unroll
      for (int kb = 0; kb < 8; ++kb) {
        const int k0 = kb * 32 + half * 8;
        const v8h* wp = (const v8h*)(w1h + n * W1S + k0);  // 16B-aligned ds
        v8h blo = wp[0], bhi = wp[2];
        v16h b = __builtin_shufflevector(blo, bhi,
            0, 1, 2, 3, 4, 5, 6, 7, 8, 9, 10, 11, 12, 13, 14, 15);
        accA = wmma_f16(afA[kb], b, accA);   // one B fragment,
        accB = wmma_f16(afB[kb], b, accB);   // two WMMAs
      }
      const float w2v = w2row[n];
#pragma unroll
      for (int r = 0; r < 8; ++r) {
        dotA[r] += fast_tanh(accA[r]) * w2v;   // layer-3 dot, fused
        dotB[r] += fast_tanh(accB[r]) * w2v;
      }
    }

    // Reduce over the 16 lanes of each C half-group, final tanh, write col i.
#pragma unroll
    for (int r = 0; r < 8; ++r) {
      float vA = dotA[r], vB = dotB[r];
      vA += __shfl_xor(vA, 1, 16);
      vA += __shfl_xor(vA, 2, 16);
      vA += __shfl_xor(vA, 4, 16);
      vA += __shfl_xor(vA, 8, 16);
      vB += __shfl_xor(vB, 1, 16);
      vB += __shfl_xor(vB, 2, 16);
      vB += __shfl_xor(vB, 4, 16);
      vB += __shfl_xor(vB, 8, 16);
      if (l16 == 0) {
        const int m0 = s0 * 16 + r + half * 8;
        const int m1 = s1 * 16 + r + half * 8;
        out[m0 * NN + i] = fast_tanh(vA);
        out[m1 * NN + i] = fast_tanh(vB);
      }
    }
  }
}

extern "C" void kernel_launch(void* const* d_in, const int* in_sizes, int n_in,
                              void* d_out, int out_size, void* d_ws, size_t ws_size,
                              hipStream_t stream) {
  (void)in_sizes; (void)n_in; (void)out_size; (void)ws_size;
  const float* x  = (const float*)d_in[0];
  const float* W0 = (const float*)d_in[1];
  const float* W1 = (const float*)d_in[2];
  const float* W2 = (const float*)d_in[3];
  float* base = (float*)d_ws;                 // 1024*256 f32 = 1 MB scratch
  float* out  = (float*)d_out;

  base_gemm_kernel<<<64, 256, 0, stream>>>(x, W0, base);

  const size_t lds_bytes = (size_t)NN * W1S * 2 + 2 * NN * sizeof(float); // ~137 KB
  mlp_masked_kernel<<<NN, 256, lds_bytes, stream>>>(x, W0, W1, W2, base, out);
}
